// MaskAttn_56607668961917
// MI455X (gfx1250) — matrix operand order repaired and checked by reading, not compiled, not run on hardware
//
#include <hip/hip_runtime.h>
#include <hip/hip_bf16.h>

#define B_ 4
#define N_ 2048
#define C_ 1024
#define H_ 16
#define HD_ 64
#define D3_ (3 * C_)
#define M_ (B_ * N_)
#define SCALE_ 0.125f  // HD^-0.5

typedef __attribute__((ext_vector_type(16))) __bf16 v16bf;
typedef __attribute__((ext_vector_type(8)))  __bf16 v8bf;
typedef __attribute__((ext_vector_type(2)))  __bf16 v2bf;
typedef __attribute__((ext_vector_type(8)))  float  v8f;
typedef __attribute__((ext_vector_type(4)))  unsigned int u32x4;
typedef __attribute__((ext_vector_type(8)))  unsigned int u32x8;

// ds_swizzle_b32 immediate XOR butterflies (group-of-32: and=0x1f, xor<<10)
#define SWZ_XOR(x, imm) \
    __int_as_float(__builtin_amdgcn_ds_swizzle(__float_as_int(x), (imm)))

// ---------------------------------------------------------------------------
// fp32 -> bf16 conversion, two elements per thread (v_cvt_pk_bf16_f32 path)
// ---------------------------------------------------------------------------
__global__ void cvt_f32_bf16(const float* __restrict__ src,
                             __bf16* __restrict__ dst, int nPairs) {
    const float2* s2 = (const float2*)src;
    v2bf* d2 = (v2bf*)dst;
    for (int i = blockIdx.x * blockDim.x + threadIdx.x; i < nPairs;
         i += gridDim.x * blockDim.x) {
        float2 f = s2[i];
        v2bf o; o[0] = (__bf16)f.x; o[1] = (__bf16)f.y;
        d2[i] = o;
    }
}

// ---------------------------------------------------------------------------
// Tensor Data Mover: 2D tile load global->LDS (CDNA5 ISA ch.8, D# descriptor)
// Group 0 (4 SGPRs): count=1 | lds_addr | global_addr | type=2
// Group 1 (8 SGPRs): data_size=2B, tensor dims, tile dims, dim0 stride
// Rows land packed contiguously in LDS (tile row-major, pitch = tileW).
// ---------------------------------------------------------------------------
__device__ inline void tdm_load_2d(unsigned ldsOff, const void* gptr,
                                   unsigned tileW, unsigned tileH,
                                   unsigned strideElems) {
    unsigned long long ga = (unsigned long long)(uintptr_t)gptr;
    u32x4 g0;
    g0[0] = 1u;                                   // count=1, user descriptor
    g0[1] = ldsOff;                               // lds_addr (bytes)
    g0[2] = (unsigned)ga;                         // global_addr[31:0]
    g0[3] = (unsigned)(ga >> 32) | 0x80000000u;   // global_addr[56:32] | type=2
    u32x8 g1;
    g1[0] = 1u << 16;                             // wg_mask=0, data_size=1 (2B)
    g1[1] = (tileW & 0xFFFFu) << 16;              // tensor_dim0 lo16
    g1[2] = (tileW >> 16) | ((tileH & 0xFFFFu) << 16);  // d0 hi | tensor_dim1 lo
    g1[3] = (tileH >> 16) | (tileW << 16);        // d1 hi | tile_dim0
    g1[4] = tileH & 0xFFFFu;                      // tile_dim1 (tile_dim2=0)
    g1[5] = strideElems;                          // tensor_dim0_stride lo32
    g1[6] = 0u;
    g1[7] = 0u;
    asm volatile("tensor_load_to_lds %0, %1" :: "s"(g0), "s"(g1) : "memory");
}

// ---------------------------------------------------------------------------
// WMMA fragment loaders (CDNA5 ISA 7.12.2 layouts, wave32)
// ---------------------------------------------------------------------------
__device__ inline v16bf load_a_frag(const __bf16* base, int ld, int mBase,
                                    int kBase, int lane) {
    int hi = lane >> 4;
    const __bf16* row = base + (size_t)(mBase + (lane & 15)) * ld + kBase + hi * 8;
    v16bf a;
#pragma unroll
    for (int j = 0; j < 8; ++j) { a[j] = row[j]; a[8 + j] = row[16 + j]; }
    return a;
}

__device__ inline v16bf load_b_frag(const __bf16* base, int ld, int nBase,
                                    int kBase, int lane) {
    int hi = lane >> 4;
    const __bf16* row = base + (size_t)(nBase + (lane & 15)) * ld + kBase + hi * 8;
    v8bf lo = *(const v8bf*)row;
    v8bf up = *(const v8bf*)(row + 16);
    return __builtin_shufflevector(lo, up, 0, 1, 2, 3, 4, 5, 6, 7, 8, 9, 10, 11, 12, 13, 14, 15);
}

__device__ inline v8f wmma_bf16(v16bf a, v16bf b, v8f c) {
    return __builtin_amdgcn_wmma_f32_16x16x32_bf16(
        false, a, false, b, (short)0, c, false, false);
}

// ---------------------------------------------------------------------------
// GEMM: out[m,d] = sum_c A[m,c] * W[d,c]   (A bf16 [M,K], W bf16 [D,K])
// Block: 256 threads (8 waves), 128x64 tile; each wave owns 32x32 (4 accums).
// K staged in 64-steps via double-buffered TDM (8 WMMA/wave between barriers).
// MODE 0: QKV projection -> scatter bf16 to [3][B,H,N,HD], Q scaled.
// MODE 1: out projection -> fp32 out[m*C_+d] + bias[d].
// ---------------------------------------------------------------------------
template <int MODE>
__global__ __launch_bounds__(256)
void gemm_bf16(const __bf16* __restrict__ A, const __bf16* __restrict__ W,
               __bf16* __restrict__ qkvOut, const float* __restrict__ bias,
               float* __restrict__ fOut, int Kdim) {
    __shared__ __align__(32) __bf16 As[2][128 * 64];   // 16 KB each
    __shared__ __align__(32) __bf16 Bs[2][64 * 64];    //  8 KB each

    const int tid  = threadIdx.x;
    const int wave = tid >> 5;
    const int lane = tid & 31;
    const int wm = wave & 3;      // 0..3 : 32-row slab of the 128-row tile
    const int wn = wave >> 2;     // 0..1 : 32-col slab of the 64-col tile
    const int m0 = blockIdx.y * 128;
    const int n0 = blockIdx.x * 64;

    v8f acc00 = {}, acc01 = {}, acc10 = {}, acc11 = {};

    const int nT = Kdim >> 6;     // 64-wide K stages
    if (wave == 0) {   // one TDM issue per block (EXEC ignored by tensor ops)
        tdm_load_2d((unsigned)(uintptr_t)&As[0][0], A + (size_t)m0 * Kdim,
                    64, 128, (unsigned)Kdim);
        tdm_load_2d((unsigned)(uintptr_t)&Bs[0][0], W + (size_t)n0 * Kdim,
                    64, 64, (unsigned)Kdim);
    }

    for (int t = 0; t < nT; ++t) {
        const int cur = t & 1;
        if (wave == 0) {
            if (t + 1 < nT) {
                const int kk = (t + 1) << 6;
                tdm_load_2d((unsigned)(uintptr_t)&As[cur ^ 1][0],
                            A + (size_t)m0 * Kdim + kk, 64, 128, (unsigned)Kdim);
                tdm_load_2d((unsigned)(uintptr_t)&Bs[cur ^ 1][0],
                            W + (size_t)n0 * Kdim + kk, 64, 64, (unsigned)Kdim);
                // tensor ops retire in order: <=2 means the current pair landed
                __builtin_amdgcn_s_wait_tensorcnt(2);
            } else {
                __builtin_amdgcn_s_wait_tensorcnt(0);
            }
        }
        __syncthreads();

#pragma unroll
        for (int ks = 0; ks < 2; ++ks) {
            const int kb = ks * 32;
            v16bf a0 = load_a_frag(As[cur], 64, wm * 32,      kb, lane);
            v16bf a1 = load_a_frag(As[cur], 64, wm * 32 + 16, kb, lane);
            v16bf b0 = load_b_frag(Bs[cur], 64, wn * 32,      kb, lane);
            v16bf b1 = load_b_frag(Bs[cur], 64, wn * 32 + 16, kb, lane);
            acc00 = wmma_bf16(a0, b0, acc00);
            acc01 = wmma_bf16(a0, b1, acc01);
            acc10 = wmma_bf16(a1, b0, acc10);
            acc11 = wmma_bf16(a1, b1, acc11);
        }

        __syncthreads();   // all waves done reading before TDM overwrites
    }

    // epilogue: C layout -> VGPR i = row (i or i+8 per half-wave), lane&15 = col
    const int hi = lane >> 4, lc = lane & 15;
#pragma unroll
    for (int mt = 0; mt < 2; ++mt) {
#pragma unroll
        for (int nt = 0; nt < 2; ++nt) {
            v8f& acc = mt ? (nt ? acc11 : acc10) : (nt ? acc01 : acc00);
#pragma unroll
            for (int i = 0; i < 8; ++i) {
                int m = m0 + wm * 32 + mt * 16 + hi * 8 + i;
                int d = n0 + wn * 32 + nt * 16 + lc;
                float val = acc[i];
                if (MODE == 0) {
                    int which = d >> 10;     // 0=Q 1=K 2=V
                    int rem   = d & (C_ - 1);
                    int hh = rem >> 6, hd = rem & (HD_ - 1);
                    int bb = m >> 11, nn = m & (N_ - 1);
                    if (which == 0) val *= SCALE_;
                    size_t idx = (size_t)which * ((size_t)B_ * H_ * N_ * HD_) +
                                 ((((size_t)bb * H_ + hh) * N_) + nn) * HD_ + hd;
                    qkvOut[idx] = (__bf16)val;
                } else {
                    fOut[(size_t)m * C_ + d] = val + bias[d];
                }
            }
        }
    }
}

// ---------------------------------------------------------------------------
// Flash attention: block = (b,h,64 queries), 4 waves x 16 rows each.
// K tiles double-buffered via TDM; V staged transposed in LDS.
// Online softmax in f32 with ds_swizzle XOR butterflies.
// ---------------------------------------------------------------------------
__global__ __launch_bounds__(128)
void flash_attn(const __bf16* __restrict__ Qb, const __bf16* __restrict__ Kb,
                const __bf16* __restrict__ Vb, const float* __restrict__ mWin,
                __bf16* __restrict__ attnOut) {
    __shared__ __align__(32) __bf16 Ks[2][64 * 64];  // [key][hd] (TDM targets)
    __shared__ __align__(32) __bf16 Vt[64 * 64];     // [hd][key] (transposed)
    __shared__ float biasS[64];
    __shared__ __align__(32) __bf16 Ps[4][16 * 64];  // per-wave P slab

    const int tid  = threadIdx.x;
    const int wave = tid >> 5;
    const int lane = tid & 31;
    const int hi = lane >> 4, lc = lane & 15;

    const int h = blockIdx.y, b = blockIdx.z;
    const int q0 = blockIdx.x * 64 + wave * 16;   // this wave's 16 query rows

    const size_t headOff = (((size_t)b * H_ + h) * N_) * HD_;
    const __bf16* Qp = Qb + headOff;
    const __bf16* Kp = Kb + headOff;
    const __bf16* Vp = Vb + headOff;

    // Q fragments (held for the whole kernel): 2 k-steps over HD=64
    v16bf qa0 = load_a_frag(Qp, HD_, q0, 0,  lane);
    v16bf qa1 = load_a_frag(Qp, HD_, q0, 32, lane);

    float mi[8], li[8];
#pragma unroll
    for (int i = 0; i < 8; ++i) { mi[i] = -1e30f; li[i] = 0.0f; }
    v8f o0 = {}, o1 = {}, o2 = {}, o3 = {};

    const int nKT = N_ / 64;
    if (wave == 0)     // prologue: DMA first K tile
        tdm_load_2d((unsigned)(uintptr_t)&Ks[0][0], Kp, 64, 64, HD_);

    for (int ti = 0; ti < nKT; ++ti) {
        const int cur = ti & 1;
        const int kt  = ti * 64;
        __syncthreads();   // previous tile fully consumed
        if (wave == 0 && ti + 1 < nKT)   // next K tile streams during compute
            tdm_load_2d((unsigned)(uintptr_t)&Ks[cur ^ 1][0],
                        Kp + (size_t)(ti + 1) * 64 * HD_, 64, 64, HD_);
        for (int u = tid; u < 64 * 64; u += 128) {
            int key = u >> 6, hd = u & 63;
            Vt[hd * 64 + key] = Vp[(size_t)(kt + key) * 64 + hd];
        }
        __builtin_prefetch(Vp + (size_t)(kt + 64) * 64, 0, 3);  // next V tile
        for (int u = tid; u < 64; u += 128)
            biasS[u] = mWin[b * N_ + kt + u] * 100.0f - 100.0f;
        if (wave == 0) {
            if (ti + 1 < nKT) __builtin_amdgcn_s_wait_tensorcnt(1);
            else              __builtin_amdgcn_s_wait_tensorcnt(0);
        }
        __syncthreads();

        // S = Q @ K^T : 4 col-tiles x 2 k-steps
        v8f s[4] = {};
#pragma unroll
        for (int t = 0; t < 4; ++t) {
            s[t] = wmma_bf16(qa0, load_b_frag(Ks[cur], 64, t * 16, 0,  lane), s[t]);
            s[t] = wmma_bf16(qa1, load_b_frag(Ks[cur], 64, t * 16, 32, lane), s[t]);
        }

        float bval[4];
#pragma unroll
        for (int t = 0; t < 4; ++t) bval[t] = biasS[t * 16 + lc];

        // online softmax: row i lives across lanes of one 16-lane half-group
        float alpha[8];
#pragma unroll
        for (int i = 0; i < 8; ++i) {
            float x = s[0][i] + bval[0];
#pragma unroll
            for (int t = 1; t < 4; ++t) x = fmaxf(x, s[t][i] + bval[t]);
            x = fmaxf(x, SWZ_XOR(x, 0x041F));   // xor 1
            x = fmaxf(x, SWZ_XOR(x, 0x081F));   // xor 2
            x = fmaxf(x, SWZ_XOR(x, 0x101F));   // xor 4
            x = fmaxf(x, SWZ_XOR(x, 0x201F));   // xor 8
            float mnew = fmaxf(mi[i], x);
            alpha[i] = __expf(mi[i] - mnew);
            mi[i] = mnew;
        }
#pragma unroll
        for (int i = 0; i < 8; ++i) {
            float rs = 0.0f;
#pragma unroll
            for (int t = 0; t < 4; ++t) {
                float p = __expf(s[t][i] + bval[t] - mi[i]);
                s[t][i] = p;
                rs += p;
            }
            rs += SWZ_XOR(rs, 0x041F);
            rs += SWZ_XOR(rs, 0x081F);
            rs += SWZ_XOR(rs, 0x101F);
            rs += SWZ_XOR(rs, 0x201F);
            li[i] = li[i] * alpha[i] + rs;
            o0[i] *= alpha[i]; o1[i] *= alpha[i];
            o2[i] *= alpha[i]; o3[i] *= alpha[i];
        }

        // reshape P (C-layout regs) -> A-fragments via wave-private LDS slab
        __bf16* Pw = Ps[wave];
#pragma unroll
        for (int t = 0; t < 4; ++t)
#pragma unroll
            for (int i = 0; i < 8; ++i)
                Pw[(i + hi * 8) * 64 + t * 16 + lc] = (__bf16)s[t][i];
        // same-wave DS ordering: compiler inserts s_wait_dscnt before reload

        // O += P @ V : contraction over 64 keys = 2 k-steps
#pragma unroll
        for (int ks = 0; ks < 2; ++ks) {
            v16bf pa = load_a_frag(Pw, 64, 0, ks * 32, lane);
            o0 = wmma_bf16(pa, load_b_frag(Vt, 64, 0,  ks * 32, lane), o0);
            o1 = wmma_bf16(pa, load_b_frag(Vt, 64, 16, ks * 32, lane), o1);
            o2 = wmma_bf16(pa, load_b_frag(Vt, 64, 32, ks * 32, lane), o2);
            o3 = wmma_bf16(pa, load_b_frag(Vt, 64, 48, ks * 32, lane), o3);
        }
    }

    // epilogue: O / l  -> attn buffer [b, n, h*HD + hd] as bf16
#pragma unroll
    for (int t = 0; t < 4; ++t) {
        v8f& o = (t == 0) ? o0 : (t == 1) ? o1 : (t == 2) ? o2 : o3;
#pragma unroll
        for (int i = 0; i < 8; ++i) {
            int row = q0 + hi * 8 + i;
            int hd  = t * 16 + lc;
            attnOut[((size_t)(b * N_ + row)) * C_ + h * HD_ + hd] =
                (__bf16)(o[i] / li[i]);
        }
    }
}

// ---------------------------------------------------------------------------
// launch
// ---------------------------------------------------------------------------
extern "C" void kernel_launch(void* const* d_in, const int* in_sizes, int n_in,
                              void* d_out, int out_size, void* d_ws, size_t ws_size,
                              hipStream_t stream) {
    const float* x      = (const float*)d_in[0];  // [B,N,C]
    const float* mWin   = (const float*)d_in[1];  // [B,N,1]
    const float* w_qkv  = (const float*)d_in[2];  // [3C,C]
    const float* w_proj = (const float*)d_in[3];  // [C,C]
    const float* b_proj = (const float*)d_in[4];  // [C]
    float* out = (float*)d_out;                   // [B,N,C] fp32

    char* ws = (char*)d_ws;
    size_t off = 0;
    auto alloc = [&](size_t bytes) { char* p = ws + off; off = (off + bytes + 255) & ~(size_t)255; return p; };
    __bf16* xb     = (__bf16*)alloc((size_t)M_ * C_ * 2);        // 16 MB
    __bf16* wqkvb  = (__bf16*)alloc((size_t)D3_ * C_ * 2);       //  6 MB
    __bf16* wprojb = (__bf16*)alloc((size_t)C_ * C_ * 2);        //  2 MB
    __bf16* qkvb   = (__bf16*)alloc((size_t)3 * M_ * C_ * 2);    // 48 MB
    __bf16* attnb  = (__bf16*)alloc((size_t)M_ * C_ * 2);        // 16 MB

    __bf16* Qb = qkvb;
    __bf16* Kb = qkvb + (size_t)M_ * C_;
    __bf16* Vb = qkvb + (size_t)2 * M_ * C_;

    // 1) fp32 -> bf16 conversions (paired -> v_cvt_pk_bf16_f32)
    cvt_f32_bf16<<<2048, 256, 0, stream>>>(x,      xb,     (M_ * C_) / 2);
    cvt_f32_bf16<<<2048, 256, 0, stream>>>(w_qkv,  wqkvb,  (D3_ * C_) / 2);
    cvt_f32_bf16<<<1024, 256, 0, stream>>>(w_proj, wprojb, (C_ * C_) / 2);

    // 2) QKV projection (Q pre-scaled), scatter to [3][B,H,N,HD]
    dim3 g1(D3_ / 64, M_ / 128);
    gemm_bf16<0><<<g1, 256, 0, stream>>>(xb, wqkvb, qkvb, nullptr, nullptr, C_);

    // 3) fused masked-softmax flash attention (double-buffered TDM K tiles)
    dim3 g2(N_ / 64, H_, B_);
    flash_attn<<<g2, 128, 0, stream>>>(Qb, Kb, Vb, mWin, attnb);

    // 4) output projection + bias -> fp32
    dim3 g3(C_ / 64, M_ / 128);
    gemm_bf16<1><<<g3, 256, 0, stream>>>(attnb, wprojb, nullptr, b_proj, out, C_);
}
